// CCP_56942676410706
// MI455X (gfx1250) — matrix operand order repaired and checked by reading, not compiled, not run on hardware
//
#include <hip/hip_runtime.h>

#define CC   256
#define HH   48
#define WW_  48
#define HWC  2304          // 48*48
#define NN   2304          // rois per level
#define PP   7
#define PP2  49
#define MM   (NN*PP2)      // 112896 rows for big GEMMs

#define LDS_STRIDE   264                    // bf16 elems per LDS row (256 + 8 pad)
#define LDS_STRIDE_B (LDS_STRIDE*2)         // 528 bytes
#define LDS_ROWS     64

typedef __bf16 bf16_t;
typedef __bf16 v16bf __attribute__((ext_vector_type(16)));
typedef __bf16 v8bf  __attribute__((ext_vector_type(8)));
typedef float  v8f   __attribute__((ext_vector_type(8)));

__device__ __forceinline__ bf16_t f2bf(float x) {
    unsigned u = __float_as_uint(x);
    u += 0x7FFFu + ((u >> 16) & 1u);         // round-to-nearest-even
    unsigned short h = (unsigned short)(u >> 16);
    bf16_t r; __builtin_memcpy(&r, &h, 2); return r;
}

__device__ __forceinline__ float clampf(float v, float lo, float hi) {
    return fminf(fmaxf(v, lo), hi);
}

// ---------------- weight f32 -> bf16 ----------------
__global__ void k_f2bf(const float* __restrict__ s, bf16_t* __restrict__ d, int n) {
    int i = blockIdx.x * blockDim.x + threadIdx.x;
    if (i < n) d[i] = f2bf(s[i]);
}

// ---------------- direct 3x3 conv, SAME, NCHW / OIHW ----------------
__global__ void k_conv3(const float* __restrict__ x, const float* __restrict__ w,
                        const float* __restrict__ b, float* __restrict__ y,
                        int Cout, int doRelu) {
    int idx = blockIdx.x * blockDim.x + threadIdx.x;
    if (idx >= Cout * HWC) return;
    int o = idx / HWC, pix = idx % HWC;
    int yy = pix / WW_, xx = pix % WW_;
    float acc = b[o];
    const float* wo = w + (size_t)o * CC * 9;
    for (int c = 0; c < CC; ++c) {
        const float* xc = x + (size_t)c * HWC;
        const float* wc = wo + c * 9;
#pragma unroll
        for (int ky = 0; ky < 3; ++ky) {
            int iy = yy + ky - 1;
            if (iy < 0 || iy >= HH) continue;
#pragma unroll
            for (int kx = 0; kx < 3; ++kx) {
                int ix = xx + kx - 1;
                if (ix < 0 || ix >= WW_) continue;
                acc = fmaf(wc[ky * 3 + kx], xc[iy * WW_ + ix], acc);
            }
        }
    }
    if (doRelu) acc = fmaxf(acc, 0.f);
    y[idx] = acc;
}

// ---------------- get_rois (two chained levels) ----------------
__global__ void k_rois(const float* __restrict__ reg,      // [8, 2304]
                       float* __restrict__ rois0, float* __restrict__ rois1,
                       int* __restrict__ cx0, int* __restrict__ cy0,
                       int* __restrict__ cx1, int* __restrict__ cy1) {
    int n = blockIdx.x * blockDim.x + threadIdx.x;
    if (n >= NN) return;
    float xc = (float)(n % WW_);
    float yc = (float)(n / WW_);
#pragma unroll
    for (int i = 0; i < 2; ++i) {
        float dx = reg[(0 * 2 + i) * HWC + n];
        float dy = reg[(1 * 2 + i) * HWC + n];
        float dw = reg[(2 * 2 + i) * HWC + n];
        float dh = reg[(3 * 2 + i) * HWC + n];
        float xcn = clampf(xc + dx * 3.0f, 0.f, 47.f);
        float ycn = clampf(yc + dy * 3.0f, 0.f, 47.f);
        float bw = 12.0f * __expf(clampf(dw, -2.f, 2.f));
        float bh = 12.0f * __expf(clampf(dh, -2.f, 2.f));
        float x1 = clampf(xcn - bw * 0.5f, 0.f, 47.f);
        float x2 = clampf(xcn + bw * 0.5f, 0.f, 47.f);
        float y1 = clampf(ycn - bh * 0.5f, 0.f, 47.f);
        float y2 = clampf(ycn + bh * 0.5f, 0.f, 47.f);
        float* rr = i ? rois1 : rois0;
        rr[n * 4 + 0] = x1; rr[n * 4 + 1] = y1;
        rr[n * 4 + 2] = x2; rr[n * 4 + 3] = y2;
        int xi = (int)clampf(rintf(xcn), 0.f, 47.f);
        int yi = (int)clampf(rintf(ycn), 0.f, 47.f);
        if (i == 0) { cx0[n] = xi; cy0[n] = yi; }
        else        { cx1[n] = xi; cy1[n] = yi; }
        xc = xcn; yc = ycn;
    }
}

// ---------------- ROI-align (+optional f32 residual) -> bf16 rows ----------
__global__ void k_roialign(const float* __restrict__ f,    // [256,48,48]
                           const float* __restrict__ rois, // [N,4]
                           const float* __restrict__ resid,// [M,256] or null
                           bf16_t* __restrict__ Xbf) {     // [M,256]
    int row = blockIdx.x;          // 0..M-1
    int c   = threadIdx.x;         // 0..255
    int n = row / PP2, p = row % PP2, py = p / PP, px = p % PP;
    float x1 = rois[n * 4 + 0], y1 = rois[n * 4 + 1];
    float x2 = rois[n * 4 + 2], y2 = rois[n * 4 + 3];
    float sx = clampf(x1 + ((px + 0.5f) / (float)PP) * (x2 - x1), 0.f, 47.f);
    float sy = clampf(y1 + ((py + 0.5f) / (float)PP) * (y2 - y1), 0.f, 47.f);
    float x0f = floorf(sx), y0f = floorf(sy);
    float fx = sx - x0f, fy = sy - y0f;
    int x0 = (int)x0f, y0 = (int)y0f;
    int x1i = min(x0 + 1, WW_ - 1), y1i = min(y0 + 1, HH - 1);
    const float* fc_ = f + (size_t)c * HWC;
    float v00 = fc_[y0 * WW_ + x0],  v01 = fc_[y0 * WW_ + x1i];
    float v10 = fc_[y1i * WW_ + x0], v11 = fc_[y1i * WW_ + x1i];
    float val = v00 * (1.f - fy) * (1.f - fx) + v01 * (1.f - fy) * fx
              + v10 * fy * (1.f - fx)        + v11 * fy * fx;
    if (resid) val += resid[(size_t)row * CC + c];
    Xbf[(size_t)row * CC + c] = f2bf(val);
}

// ---- async-stage a 64x256 bf16 W panel into LDS (padded rows) -------------
// Each of 256 threads copies 16B chunks; VDST VGPR carries the wave-relative
// LDS byte offset (= low 32 bits of the generic shared pointer).
__device__ __forceinline__ void stage_w(const char* __restrict__ wsrc,
                                        unsigned ldsBase, int tid) {
    int rr = tid >> 5, ch = tid & 31;        // 8 rows/iter, 32 x 16B chunks/row
#pragma unroll
    for (int it = 0; it < 8; ++it) {
        int row = it * 8 + rr;
        unsigned lofs = ldsBase + row * LDS_STRIDE_B + ch * 16;
        const char* g = wsrc + row * 512 + ch * 16;
        asm volatile("global_load_async_to_lds_b128 %0, %1, off"
                     :: "v"(lofs), "v"(g) : "memory");
    }
#if __has_builtin(__builtin_amdgcn_s_wait_asynccnt)
    __builtin_amdgcn_s_wait_asynccnt(0);
#else
    asm volatile("s_wait_asynccnt 0x0" ::: "memory");
#endif
}

// ---- per-wave 16x64 WMMA pass over K=256 with B from LDS ------------------
__device__ __forceinline__ void wmma_pass(const bf16_t* __restrict__ aRow,   // +hs*8 applied
                                          const bf16_t* __restrict__ lb,     // lw + l15*STRIDE + hs*8
                                          v8f acc[4]) {
#pragma unroll
    for (int kb = 0; kb < CC; kb += 32) {
        union { v16bf v; v8bf h[2]; } a;
        a.h[0] = *(const v8bf*)(aRow + kb);          // K = kb + hs*8 + [0..7]
        a.h[1] = *(const v8bf*)(aRow + kb + 16);     // K = kb + hs*8 + [16..23]
#pragma unroll
        for (int j = 0; j < 4; ++j) {
            const bf16_t* p = lb + (size_t)j * (16 * LDS_STRIDE) + kb;
            union { v16bf v; v8bf h[2]; } b;
            b.h[0] = *(const v8bf*)(p);              // K = kb + hs*8 + [0..7]
            b.h[1] = *(const v8bf*)(p + 16);         // K = kb + hs*8 + [16..23]
            acc[j] = __builtin_amdgcn_wmma_f32_16x16x32_bf16(
                false, a.v, false, b.v, (short)0, acc[j], false, false);
        }
    }
}

// Y[row,col] = act( sum_k A[row,k]*W[col,k] + bias[col] )
// Block = 8 waves = 128(M) x 64(N) macro-tile; W panel staged in LDS.
// Requires M % 128 == 0 (true for 112896 and 2304).
__global__ void __launch_bounds__(256)
k_gemm(const bf16_t* __restrict__ A, const bf16_t* __restrict__ Wb,
       const float* __restrict__ bias,
       float* __restrict__ Yf, bf16_t* __restrict__ Ybf,
       int M, int doRelu) {
    __shared__ bf16_t lw[LDS_ROWS * LDS_STRIDE];
    const int nGroup = blockIdx.x & 3;
    const int mBlock = blockIdx.x >> 2;
    stage_w((const char*)(Wb + (size_t)nGroup * 64 * CC),
            (unsigned)(uintptr_t)lw, threadIdx.x);
    __syncthreads();

    const int wave = threadIdx.x >> 5, lane = threadIdx.x & 31;
    const int hs = lane >> 4, l15 = lane & 15;
    const int mTile = mBlock * 8 + wave;
    const bf16_t* aRow = A + (size_t)(mTile * 16 + l15) * CC + hs * 8;
    const bf16_t* lb = lw + (size_t)l15 * LDS_STRIDE + hs * 8;
    v8f acc[4] = {};
    wmma_pass(aRow, lb, acc);

    const int rowBase = mTile * 16 + hs * 8;
#pragma unroll
    for (int j = 0; j < 4; ++j) {
        const int col = nGroup * 64 + j * 16 + l15;
        const float bv = bias[col];
#pragma unroll
        for (int r = 0; r < 8; ++r) {
            float v = acc[j][r] + bv;
            if (doRelu) v = fmaxf(v, 0.f);
            size_t idx = (size_t)(rowBase + r) * CC + col;
            if (Yf)  Yf[idx]  = v;
            if (Ybf) Ybf[idx] = f2bf(v);
        }
    }
}

// a2a: Y = relu( A0*W0^T + A1*W1^T + A2*W2^T + bias ), restaging LDS per term
__global__ void __launch_bounds__(256)
k_gemm3(const bf16_t* __restrict__ A0, const bf16_t* __restrict__ A1,
        const bf16_t* __restrict__ A2,
        const bf16_t* __restrict__ W0, const bf16_t* __restrict__ W1,
        const bf16_t* __restrict__ W2,
        const float* __restrict__ bias, float* __restrict__ Yf, int M) {
    __shared__ bf16_t lw[LDS_ROWS * LDS_STRIDE];
    const int nGroup = blockIdx.x & 3;
    const int mBlock = blockIdx.x >> 2;
    const int wave = threadIdx.x >> 5, lane = threadIdx.x & 31;
    const int hs = lane >> 4, l15 = lane & 15;
    const int mTile = mBlock * 8 + wave;
    const size_t aOff = (size_t)(mTile * 16 + l15) * CC + hs * 8;
    const bf16_t* lb = lw + (size_t)l15 * LDS_STRIDE + hs * 8;
    const unsigned ldsBase = (unsigned)(uintptr_t)lw;
    const bf16_t* As[3] = {A0, A1, A2};
    const bf16_t* Ws[3] = {W0, W1, W2};
    v8f acc[4] = {};
#pragma unroll 1
    for (int s = 0; s < 3; ++s) {
        stage_w((const char*)(Ws[s] + (size_t)nGroup * 64 * CC), ldsBase, threadIdx.x);
        __syncthreads();
        wmma_pass(As[s] + aOff, lb, acc);
        __syncthreads();                 // all reads done before restage
    }
    const int rowBase = mTile * 16 + hs * 8;
#pragma unroll
    for (int j = 0; j < 4; ++j) {
        const int col = nGroup * 64 + j * 16 + l15;
        const float bv = bias[col];
#pragma unroll
        for (int r = 0; r < 8; ++r) {
            float v = fmaxf(acc[j][r] + bv, 0.f);
            Yf[(size_t)(rowBase + r) * CC + col] = v;
        }
    }
}

// ---------------- mean over 49 samples per roi ----------------
__global__ void k_pool(const float* __restrict__ rf, float* __restrict__ outf,
                       bf16_t* __restrict__ outbf) {
    int idx = blockIdx.x * blockDim.x + threadIdx.x;   // n*256+c
    if (idx >= NN * CC) return;
    int n = idx >> 8, c = idx & 255;
    const float* p = rf + (size_t)n * PP2 * CC + c;
    float s = 0.f;
#pragma unroll 7
    for (int i = 0; i < PP2; ++i) s += p[(size_t)i * CC];
    s *= (1.f / 49.f);
    outf[idx] = s;
    if (outbf) outbf[idx] = f2bf(s);
}

__global__ void k_copy(float* __restrict__ d, const float* __restrict__ s, int n) {
    int i = blockIdx.x * blockDim.x + threadIdx.x;
    if (i < n) d[i] = s[i];
}

// dst is channel-major [256,48,48]; usp is [N,256]
__global__ void k_scatter(float* __restrict__ dst, const float* __restrict__ usp,
                          const int* __restrict__ cx, const int* __restrict__ cy) {
    int idx = blockIdx.x * blockDim.x + threadIdx.x;   // n*256+c
    if (idx >= NN * CC) return;
    int n = idx >> 8, c = idx & 255;
    int pix = cy[n] * WW_ + cx[n];
    atomicAdd(dst + (size_t)c * HWC + pix, usp[idx]);
}

// out[o,y,x] = outvec[(y*48+x), o]
__global__ void k_assemble(const float* __restrict__ outvec, float* __restrict__ out) {
    int idx = blockIdx.x * blockDim.x + threadIdx.x;
    if (idx >= CC * HWC) return;
    int o = idx / HWC, pix = idx % HWC;
    out[idx] = outvec[(size_t)pix * CC + o];
}

extern "C" void kernel_launch(void* const* d_in, const int* in_sizes, int n_in,
                              void* d_out, int out_size, void* d_ws, size_t ws_size,
                              hipStream_t stream) {
    const float* fa      = (const float*)d_in[0];
    // d_in[1],[2] = convB (dead code in reference) — skipped
    const float* convC_w = (const float*)d_in[3];
    const float* convC_b = (const float*)d_in[4];
    const float* reg_w   = (const float*)d_in[5];
    const float* reg_b   = (const float*)d_in[6];
    const float* p2a_w   = (const float*)d_in[7];
    const float* p2a_b   = (const float*)d_in[8];
    const float* l2h_w   = (const float*)d_in[9];
    const float* l2h_b   = (const float*)d_in[10];
    const float* h2l_w   = (const float*)d_in[11];
    const float* h2l_b   = (const float*)d_in[12];
    const float* a2a_w1  = (const float*)d_in[13];
    const float* a2a_w2  = (const float*)d_in[14];
    const float* a2a_w3  = (const float*)d_in[15];
    const float* a2a_b   = (const float*)d_in[16];
    const float* a2p_w   = (const float*)d_in[17];
    const float* a2p_b   = (const float*)d_in[18];
    float* out = (float*)d_out;                      // [512,48,48]

    char* ws = (char*)d_ws;
    size_t off = 0;
    auto alloc = [&](size_t b) -> char* {
        char* p = ws + off; off += (b + 255) & ~(size_t)255; return p;
    };
    bf16_t* B1 = (bf16_t*)alloc((size_t)MM * CC * 2);
    bf16_t* B2 = (bf16_t*)alloc((size_t)MM * CC * 2);
    bf16_t* B3 = (bf16_t*)alloc((size_t)MM * CC * 2);
    float*  F1 = (float*) alloc((size_t)MM * CC * 4);
    float*  fc = (float*) alloc((size_t)HWC * CC * 4);
    float*  fb = (float*) alloc((size_t)HWC * CC * 4);
    float*  regB  = (float*)alloc((size_t)8 * HWC * 4);
    float*  rois0 = (float*)alloc((size_t)NN * 4 * 4);
    float*  rois1 = (float*)alloc((size_t)NN * 4 * 4);
    int* cx0 = (int*)alloc((size_t)NN * 4); int* cy0 = (int*)alloc((size_t)NN * 4);
    int* cx1 = (int*)alloc((size_t)NN * 4); int* cy1 = (int*)alloc((size_t)NN * 4);
    float* usp0 = (float*)alloc((size_t)NN * CC * 4);
    float* usp1 = (float*)alloc((size_t)NN * CC * 4);
    float* pooledf  = (float*) alloc((size_t)NN * CC * 4);
    bf16_t* pooledbf = (bf16_t*)alloc((size_t)NN * CC * 2);
    float* outvec = (float*)alloc((size_t)NN * CC * 4);
    const int wN = CC * CC;
    bf16_t* w_p2a  = (bf16_t*)alloc((size_t)wN * 2);
    bf16_t* w_l2h  = (bf16_t*)alloc((size_t)wN * 2);
    bf16_t* w_h2l  = (bf16_t*)alloc((size_t)wN * 2);
    bf16_t* w_a2a1 = (bf16_t*)alloc((size_t)wN * 2);
    bf16_t* w_a2a2 = (bf16_t*)alloc((size_t)wN * 2);
    bf16_t* w_a2a3 = (bf16_t*)alloc((size_t)wN * 2);
    bf16_t* w_a2p  = (bf16_t*)alloc((size_t)wN * 2);

    const int T = 256;
    const int gw = (wN + T - 1) / T;
    k_f2bf<<<gw, T, 0, stream>>>(p2a_w,  w_p2a,  wN);
    k_f2bf<<<gw, T, 0, stream>>>(l2h_w,  w_l2h,  wN);
    k_f2bf<<<gw, T, 0, stream>>>(h2l_w,  w_h2l,  wN);
    k_f2bf<<<gw, T, 0, stream>>>(a2a_w1, w_a2a1, wN);
    k_f2bf<<<gw, T, 0, stream>>>(a2a_w2, w_a2a2, wN);
    k_f2bf<<<gw, T, 0, stream>>>(a2a_w3, w_a2a3, wN);
    k_f2bf<<<gw, T, 0, stream>>>(a2p_w,  w_a2p,  wN);

    // fc = relu(conv3x3(fa, convC)), reg = conv3x3(fa, reg_w)   (convB dead)
    k_conv3<<<(CC * HWC + T - 1) / T, T, 0, stream>>>(fa, convC_w, convC_b, fc, CC, 1);
    k_conv3<<<(8 * HWC + T - 1) / T,  T, 0, stream>>>(fa, reg_w,  reg_b,  regB, 8, 0);

    k_rois<<<(NN + T - 1) / T, T, 0, stream>>>(regB, rois0, rois1, cx0, cy0, cx1, cy1);

    const int gemmBlk  = (MM / 128) * 4;   // 3528 blocks, 128x64 macro-tiles
    const int gemmBlkS = (NN / 128) * 4;   // 72 blocks
    const int pg = (NN * CC + T - 1) / T;
    const int cg = (CC * HWC + T - 1) / T;

    // rf1 = relu(RA(fa,rois0) x p2a^T + b); usp0 = pool(rf1)
    k_roialign<<<MM, T, 0, stream>>>(fa, rois0, nullptr, B1);
    k_gemm<<<gemmBlk, T, 0, stream>>>(B1, w_p2a, p2a_b, F1, nullptr, MM, 1);
    k_pool<<<pg, T, 0, stream>>>(F1, usp0, nullptr);

    // rf2 = relu((rf1 + RA(fa,rois1)) x l2h^T + b); usp1 = pool(rf2)
    k_roialign<<<MM, T, 0, stream>>>(fa, rois1, F1, B1);
    k_gemm<<<gemmBlk, T, 0, stream>>>(B1, w_l2h, l2h_b, F1, B2, MM, 1);
    k_pool<<<pg, T, 0, stream>>>(F1, usp1, nullptr);

    // fb = fa + scatter(usp0@c0) + scatter(usp1@c1)
    k_copy<<<cg, T, 0, stream>>>(fb, fa, CC * HWC);
    k_scatter<<<pg, T, 0, stream>>>(fb, usp0, cx0, cy0);
    k_scatter<<<pg, T, 0, stream>>>(fb, usp1, cx1, cy1);

    // rf3 = relu(rf2*a2a1^T + RA(fb,rois1)*a2a2^T + RA(fc,rois1)*a2a3^T + b)
    k_roialign<<<MM, T, 0, stream>>>(fb, rois1, nullptr, B1);
    k_roialign<<<MM, T, 0, stream>>>(fc, rois1, nullptr, B3);
    k_gemm3<<<gemmBlk, T, 0, stream>>>(B2, B1, B3, w_a2a1, w_a2a2, w_a2a3, a2a_b, F1, MM);

    // rf4 = relu((rf3 + RA(fc,rois0)) x h2l^T + b); pooled = pool(rf4)
    k_roialign<<<MM, T, 0, stream>>>(fc, rois0, F1, B1);
    k_gemm<<<gemmBlk, T, 0, stream>>>(B1, w_h2l, h2l_b, F1, nullptr, MM, 1);
    k_pool<<<pg, T, 0, stream>>>(F1, pooledf, pooledbf);

    // out_vec = relu(pooled x a2p^T + b)   (M = 2304)
    k_gemm<<<gemmBlkS, T, 0, stream>>>(pooledbf, w_a2p, a2p_b, outvec, nullptr, NN, 1);

    // d_out = concat(fa, out + scatter(pooled@c0) + scatter(pooled@c1))
    k_copy<<<cg, T, 0, stream>>>(out, fa, CC * HWC);
    float* out2 = out + (size_t)CC * HWC;
    k_assemble<<<cg, T, 0, stream>>>(outvec, out2);
    k_scatter<<<pg, T, 0, stream>>>(out2, pooledf, cx0, cy0);
    k_scatter<<<pg, T, 0, stream>>>(out2, pooledf, cx1, cy1);
}
